// Attend_12086037971467
// MI455X (gfx1250) — matrix operand order repaired and checked by reading, not compiled, not run on hardware
//
#include <hip/hip_runtime.h>
#include <math.h>

typedef __attribute__((ext_vector_type(16))) _Float16 v16h;
typedef __attribute__((ext_vector_type(4)))  _Float16 v4h;
typedef __attribute__((ext_vector_type(8)))  float    v8f;
typedef __attribute__((ext_vector_type(4)))  unsigned int v4u;
typedef __attribute__((ext_vector_type(8)))  unsigned int v8u;

#define BH    32      // batch * heads
#define SEQ   2048
#define DH    64
#define JT    64      // keys per iteration (4 x 16-wide score tiles)
#define WAVES 4       // waves per block; each wave owns 16 query rows

// (1/sqrt(64)) * log2(e): base-2 softmax, scale folded into Q conversion
#define QSCALE 0.1803368801111204f

__device__ __forceinline__ float fexp2(float x) {
#if defined(__has_builtin)
#if __has_builtin(__builtin_amdgcn_exp2f)
    return __builtin_amdgcn_exp2f(x);
#else
    return exp2f(x);
#endif
#else
    return exp2f(x);
#endif
}

__device__ __forceinline__ unsigned int rfl(unsigned int x) {
    return (unsigned int)__builtin_amdgcn_readfirstlane((int)x);
}

// ---------------------------------------------------------------------------
// Pass 1: K -> f16 row-major [bh][s][d]; V -> f16 transposed [bh][d][s]
// ---------------------------------------------------------------------------
__global__ __launch_bounds__(256)
void cvt_kv(const float* __restrict__ k, const float* __restrict__ v,
            _Float16* __restrict__ kh, _Float16* __restrict__ vt) {
    const size_t idx4 = ((size_t)blockIdx.x * 256 + threadIdx.x) * 4;
    const float4 kq = *(const float4*)&k[idx4];
    const float4 vq = *(const float4*)&v[idx4];
    v4h khv;
    khv[0] = (_Float16)kq.x; khv[1] = (_Float16)kq.y;
    khv[2] = (_Float16)kq.z; khv[3] = (_Float16)kq.w;
    *(v4h*)&kh[idx4] = khv;
    const int    d  = (int)(idx4 & (DH - 1));
    const size_t t  = idx4 >> 6;
    const int    s  = (int)(t & (SEQ - 1));
    const size_t hh = t >> 11;
    _Float16* vrow = vt + (hh * DH + d) * SEQ + s;
    vrow[0 * SEQ] = (_Float16)vq.x;
    vrow[1 * SEQ] = (_Float16)vq.y;
    vrow[2 * SEQ] = (_Float16)vq.z;
    vrow[3 * SEQ] = (_Float16)vq.w;
}

// ---------------------------------------------------------------------------
// Pass 2: flash attention. USE_TDM=true: double-buffered tensor_load_to_lds
// staging (per-wave quarter tiles, DMA of tile t+1 overlaps compute of t).
// USE_TDM=false: self-staging fallback (single buffer).
// ---------------------------------------------------------------------------
template <bool USE_TDM>
__global__ __launch_bounds__(WAVES * 32)
void fa_fwd_wmma(const float* __restrict__ q,
                 const float* __restrict__ kf,
                 const float* __restrict__ vf,
                 const _Float16* __restrict__ kh,
                 const _Float16* __restrict__ vt,
                 float* __restrict__ o) {
    __shared__ __align__(32) _Float16 Ks[2][JT * DH];       // [buf][key][d], 16 KB
    __shared__ __align__(32) _Float16 VT[2][DH * JT];       // [buf][d][key], 16 KB
    __shared__ __align__(32) _Float16 Pb[WAVES * 16 * JT];  // per-wave P,    8 KB

    const int tid  = threadIdx.x;
    const int wv   = tid >> 5;
    const int lane = tid & 31;
    const int hl   = lane >> 4;
    const int lq   = lane & 15;

    const int nRowTiles = SEQ / (16 * WAVES);
    const int bh = blockIdx.x / nRowTiles;
    const int rt = blockIdx.x % nRowTiles;
    const int i0 = rt * (16 * WAVES) + wv * 16;

    const float* qb = q + (size_t)bh * SEQ * DH;
    float*       ob = o + (size_t)bh * SEQ * DH;

    // ---- Q A-fragments (16x32 each), loaded once, QSCALE folded in
    const float* qrow = qb + (size_t)(i0 + lq) * DH;
    v16h aq0, aq1;
#pragma unroll
    for (int e = 0; e < 8; ++e) {
        aq0[e]     = (_Float16)(qrow[hl * 16 + e]      * QSCALE);
        aq0[e + 8] = (_Float16)(qrow[8 + hl * 16 + e]  * QSCALE);
        aq1[e]     = (_Float16)(qrow[32 + hl * 16 + e] * QSCALE);
        aq1[e + 8] = (_Float16)(qrow[40 + hl * 16 + e] * QSCALE);
    }

    // ---- TDM setup: per-wave quarter tiles; group1 descriptors loop-invariant
    unsigned long long kgaBase = 0, vgaBase = 0;
    unsigned int ldsKq[2] = {0, 0}, ldsVq[2] = {0, 0};
    v8u g1k = {}, g1v = {};
    if constexpr (USE_TDM) {
        kgaBase = (unsigned long long)(uintptr_t)(kh + (size_t)bh * SEQ * DH)
                + (unsigned long long)(16 * wv * DH) * 2;
        vgaBase = (unsigned long long)(uintptr_t)(vt + (size_t)bh * DH * SEQ
                + (size_t)(16 * wv) * SEQ);
        ldsKq[0] = (unsigned int)(uintptr_t)&Ks[0][0] + (unsigned int)(16 * wv * DH * 2);
        ldsKq[1] = (unsigned int)(uintptr_t)&Ks[1][0] + (unsigned int)(16 * wv * DH * 2);
        ldsVq[0] = (unsigned int)(uintptr_t)&VT[0][0] + (unsigned int)(16 * wv * JT * 2);
        ldsVq[1] = (unsigned int)(uintptr_t)&VT[1][0] + (unsigned int)(16 * wv * JT * 2);
        // K quarter: 1-D, 1024 f16 elements, contiguous
        g1k[0] = rfl(1u << 16);                       // data_size = 2 bytes
        g1k[1] = rfl((1024u & 0xFFFFu) << 16);        // tensor_dim0 lo
        g1k[2] = rfl((1024u >> 16) | (1u << 16));     // tensor_dim0 hi | tensor_dim1=1
        g1k[3] = rfl(1024u << 16);                    // tile_dim0 = 1024
        g1k[4] = rfl(0u);
        g1k[5] = rfl(1024u);                          // tensor_dim0_stride
        g1k[6] = rfl(0u);
        g1k[7] = rfl(0u);
        // V quarter: 2-D, tile 64 x 16, row stride SEQ
        g1v[0] = rfl(1u << 16);
        g1v[1] = rfl(((unsigned)SEQ & 0xFFFFu) << 16);
        g1v[2] = rfl(((unsigned)SEQ >> 16) | (16u << 16));
        g1v[3] = rfl(64u << 16);                      // tile_dim0 = 64
        g1v[4] = rfl(16u);                            // tile_dim1 = 16
        g1v[5] = rfl((unsigned)SEQ);                  // tensor_dim0_stride
        g1v[6] = rfl(0u);
        g1v[7] = rfl(0u);
    }

    auto issueTDM = [&](int j, int buf) {
        const unsigned long long kga = kgaBase + (unsigned long long)j * (DH * 2);
        const unsigned long long vga = vgaBase + (unsigned long long)j * 2;
        v4u g0k, g0v;
        g0k[0] = rfl(1u);
        g0k[1] = rfl(ldsKq[buf]);
        g0k[2] = rfl((unsigned int)kga);
        g0k[3] = rfl((unsigned int)(kga >> 32) | 0x80000000u);  // type=2
        g0v[0] = rfl(1u);
        g0v[1] = rfl(ldsVq[buf]);
        g0v[2] = rfl((unsigned int)vga);
        g0v[3] = rfl((unsigned int)(vga >> 32) | 0x80000000u);
        asm volatile("tensor_load_to_lds %0, %1" :: "s"(g0k), "s"(g1k) : "memory");
        asm volatile("tensor_load_to_lds %0, %1" :: "s"(g0v), "s"(g1v) : "memory");
    };

    v8f acc0 = {}, acc1 = {}, acc2 = {}, acc3 = {};
    float mrow[8], srow[8];   // srow = per-lane partial sums (reduced at end)
#pragma unroll
    for (int r = 0; r < 8; ++r) { mrow[r] = -INFINITY; srow[r] = 0.0f; }

    _Float16* Pw = &Pb[wv * 16 * JT];

    if constexpr (USE_TDM) issueTDM(0, 0);   // pipeline prologue

    for (int j0 = 0; j0 < SEQ; j0 += JT) {
        const int cb = USE_TDM ? ((j0 / JT) & 1) : 0;
        if constexpr (USE_TDM) {
            // issue next tile into the other buffer, then wait for current
            // tile's 2 ops (tensor ops complete in order per wave, so
            // TENSORcnt<=2 leaves only the newly issued pair outstanding)
            if (j0 + JT < SEQ) {
                issueTDM(j0 + JT, cb ^ 1);
                __builtin_amdgcn_s_wait_tensorcnt(2);
            } else {
                __builtin_amdgcn_s_wait_tensorcnt(0);
            }
        } else {
            const float* kb = kf + (size_t)bh * SEQ * DH;
            const float* vb = vf + (size_t)bh * SEQ * DH;
#pragma unroll
            for (int it = 0; it < (JT * DH) / (WAVES * 32 * 4); ++it) {
                const int idx4 = tid + it * (WAVES * 32);
                const int row  = idx4 >> 4;
                const int col  = (idx4 & 15) << 2;
                const float4 kq = *(const float4*)&kb[(size_t)(j0 + row) * DH + col];
                const float4 vq = *(const float4*)&vb[(size_t)(j0 + row) * DH + col];
                v4h khv;
                khv[0] = (_Float16)kq.x; khv[1] = (_Float16)kq.y;
                khv[2] = (_Float16)kq.z; khv[3] = (_Float16)kq.w;
                *(v4h*)&Ks[0][row * DH + col] = khv;
                VT[0][(col + 0) * JT + row] = (_Float16)vq.x;
                VT[0][(col + 1) * JT + row] = (_Float16)vq.y;
                VT[0][(col + 2) * JT + row] = (_Float16)vq.z;
                VT[0][(col + 3) * JT + row] = (_Float16)vq.w;
            }
        }
        __syncthreads();   // all waves' quarter-tiles for this buffer are in LDS

        const _Float16* KsC = &Ks[cb][0];
        const _Float16* VTC = &VT[cb][0];

        // ---- S = Q*K^T : four 16x16 tiles (64 keys), 8 WMMAs
        v8f s0 = {}, s1 = {}, s2 = {}, s3 = {};
        {
            const _Float16* kr0 = &KsC[(0 * 16 + lq) * DH + hl * 16];
            const _Float16* kr1 = &KsC[(1 * 16 + lq) * DH + hl * 16];
            const _Float16* kr2 = &KsC[(2 * 16 + lq) * DH + hl * 16];
            const _Float16* kr3 = &KsC[(3 * 16 + lq) * DH + hl * 16];
            s0 = __builtin_amdgcn_wmma_f32_16x16x32_f16(false, aq0, false, *(const v16h*)(kr0),      (short)0, s0, false, false);
            s0 = __builtin_amdgcn_wmma_f32_16x16x32_f16(false, aq1, false, *(const v16h*)(kr0 + 32), (short)0, s0, false, false);
            s1 = __builtin_amdgcn_wmma_f32_16x16x32_f16(false, aq0, false, *(const v16h*)(kr1),      (short)0, s1, false, false);
            s1 = __builtin_amdgcn_wmma_f32_16x16x32_f16(false, aq1, false, *(const v16h*)(kr1 + 32), (short)0, s1, false, false);
            s2 = __builtin_amdgcn_wmma_f32_16x16x32_f16(false, aq0, false, *(const v16h*)(kr2),      (short)0, s2, false, false);
            s2 = __builtin_amdgcn_wmma_f32_16x16x32_f16(false, aq1, false, *(const v16h*)(kr2 + 32), (short)0, s2, false, false);
            s3 = __builtin_amdgcn_wmma_f32_16x16x32_f16(false, aq0, false, *(const v16h*)(kr3),      (short)0, s3, false, false);
            s3 = __builtin_amdgcn_wmma_f32_16x16x32_f16(false, aq1, false, *(const v16h*)(kr3 + 32), (short)0, s3, false, false);
        }

        // ---- row max (only per-tile cross-lane reduction)
        float mnew[8];
        int upd = 0;
#pragma unroll
        for (int r = 0; r < 8; ++r) {
            float rmax = fmaxf(fmaxf(s0[r], s1[r]), fmaxf(s2[r], s3[r]));
#pragma unroll
            for (int msk = 1; msk < 16; msk <<= 1)
                rmax = fmaxf(rmax, __shfl_xor(rmax, msk, 32));
            mnew[r] = fmaxf(mrow[r], rmax);
            upd |= (mnew[r] > mrow[r]) ? 1 : 0;
        }
        if (__any(upd)) {   // rescale only when some row max grew
#pragma unroll
            for (int r = 0; r < 8; ++r) {
                const float alpha = fexp2(mrow[r] - mnew[r]);
                mrow[r] = mnew[r];
                srow[r] *= alpha;
                acc0[r] *= alpha; acc1[r] *= alpha;
                acc2[r] *= alpha; acc3[r] *= alpha;
            }
        }

        // ---- probabilities: per-lane partial sums, C-layout -> row-major LDS
#pragma unroll
        for (int r = 0; r < 8; ++r) {
            const float m = mrow[r];
            const float p0 = fexp2(s0[r] - m);
            const float p1 = fexp2(s1[r] - m);
            const float p2 = fexp2(s2[r] - m);
            const float p3 = fexp2(s3[r] - m);
            srow[r] += (p0 + p1) + (p2 + p3);
            _Float16* pr = &Pw[(r + hl * 8) * JT + lq];
            pr[0]  = (_Float16)p0;
            pr[16] = (_Float16)p1;
            pr[32] = (_Float16)p2;
            pr[48] = (_Float16)p3;
        }

        // ---- reload P as two 16x32 A-fragments (same-wave DS is in-order)
        v16h pa0, pa1;
#pragma unroll
        for (int e = 0; e < 8; ++e) {
            const _Float16* pr = &Pw[lq * JT + hl * 16 + e];
            pa0[e]     = pr[0];
            pa0[e + 8] = pr[8];
            pa1[e]     = pr[32];
            pa1[e + 8] = pr[40];
        }

        // ---- O += P*V : 4 D-tiles x 2 K-chunks = 8 WMMAs
        {
            const _Float16* vr0 = &VTC[(0 * 16 + lq) * JT + hl * 16];
            const _Float16* vr1 = &VTC[(1 * 16 + lq) * JT + hl * 16];
            const _Float16* vr2 = &VTC[(2 * 16 + lq) * JT + hl * 16];
            const _Float16* vr3 = &VTC[(3 * 16 + lq) * JT + hl * 16];
            acc0 = __builtin_amdgcn_wmma_f32_16x16x32_f16(false, pa0, false, *(const v16h*)(vr0),      (short)0, acc0, false, false);
            acc0 = __builtin_amdgcn_wmma_f32_16x16x32_f16(false, pa1, false, *(const v16h*)(vr0 + 32), (short)0, acc0, false, false);
            acc1 = __builtin_amdgcn_wmma_f32_16x16x32_f16(false, pa0, false, *(const v16h*)(vr1),      (short)0, acc1, false, false);
            acc1 = __builtin_amdgcn_wmma_f32_16x16x32_f16(false, pa1, false, *(const v16h*)(vr1 + 32), (short)0, acc1, false, false);
            acc2 = __builtin_amdgcn_wmma_f32_16x16x32_f16(false, pa0, false, *(const v16h*)(vr2),      (short)0, acc2, false, false);
            acc2 = __builtin_amdgcn_wmma_f32_16x16x32_f16(false, pa1, false, *(const v16h*)(vr2 + 32), (short)0, acc2, false, false);
            acc3 = __builtin_amdgcn_wmma_f32_16x16x32_f16(false, pa0, false, *(const v16h*)(vr3),      (short)0, acc3, false, false);
            acc3 = __builtin_amdgcn_wmma_f32_16x16x32_f16(false, pa1, false, *(const v16h*)(vr3 + 32), (short)0, acc3, false, false);
        }
        __syncthreads();   // all reads of this buffer done -> re-usable
    }

    // ---- epilogue: one row-sum reduction, normalize, store
#pragma unroll
    for (int r = 0; r < 8; ++r) {
        float s = srow[r];
#pragma unroll
        for (int msk = 1; msk < 16; msk <<= 1)
            s += __shfl_xor(s, msk, 32);
        const float inv = 1.0f / s;
        const int row = i0 + r + hl * 8;
        ob[(size_t)row * DH + 0 * 16 + lq] = acc0[r] * inv;
        ob[(size_t)row * DH + 1 * 16 + lq] = acc1[r] * inv;
        ob[(size_t)row * DH + 2 * 16 + lq] = acc2[r] * inv;
        ob[(size_t)row * DH + 3 * 16 + lq] = acc3[r] * inv;
    }
}

extern "C" void kernel_launch(void* const* d_in, const int* in_sizes, int n_in,
                              void* d_out, int out_size, void* d_ws, size_t ws_size,
                              hipStream_t stream) {
    (void)in_sizes; (void)n_in; (void)out_size;
    const float* q = (const float*)d_in[0];
    const float* k = (const float*)d_in[1];
    const float* v = (const float*)d_in[2];
    float* o = (float*)d_out;

    const size_t elems = (size_t)BH * SEQ * DH;
    const size_t need  = 2 * elems * sizeof(_Float16);   // Kh + VT workspace

    const int rowTiles = SEQ / (16 * WAVES);
    dim3 grid(BH * rowTiles);
    dim3 block(WAVES * 32);

    if (ws_size >= need) {
        _Float16* kh = (_Float16*)d_ws;
        _Float16* vt = kh + elems;
        cvt_kv<<<dim3((unsigned)(elems / (256 * 4))), dim3(256), 0, stream>>>(k, v, kh, vt);
        fa_fwd_wmma<true><<<grid, block, 0, stream>>>(q, k, v, kh, vt, o);
    } else {
        fa_fwd_wmma<false><<<grid, block, 0, stream>>>(q, k, v, nullptr, nullptr, o);
    }
}
